// TransformerEncoderLayer_51196010169061
// MI455X (gfx1250) — matrix operand order repaired and checked by reading, not compiled, not run on hardware
//
#include <hip/hip_runtime.h>
#include <cstdint>
#include <cstddef>

// ---------------------------------------------------------------------------
// Types for CDNA5 WMMA (wave32)
// ---------------------------------------------------------------------------
typedef __bf16 bf16;
typedef __attribute__((ext_vector_type(16))) __bf16 v16bf;
typedef __attribute__((ext_vector_type(8)))  __bf16 v8bf;
typedef __attribute__((ext_vector_type(8)))  float  v8f;
typedef __attribute__((ext_vector_type(4)))  float  f32x4;

union Frag { v16bf v; v8bf h[2]; };

constexpr int BSZ = 2, SEQ = 2048, DM = 1024, DFF = 4096, NH = 16, DH = 64;
constexpr int MR  = BSZ * SEQ;   // 4096 rows of activations

__device__ __forceinline__ v8f wmma_bf16(const Frag& a, const Frag& b, v8f c) {
    return __builtin_amdgcn_wmma_f32_16x16x32_bf16(
        /*neg_a=*/false, a.v, /*neg_b=*/false, b.v,
        /*c_mod=*/(short)0, c, /*reuse_a=*/false, /*reuse_b=*/false);
}

// ---------------------------------------------------------------------------
// fp32 -> bf16 convert
// ---------------------------------------------------------------------------
__global__ __launch_bounds__(256) void f32_to_bf16_kernel(
    const float* __restrict__ in, bf16* __restrict__ out, size_t n)
{
    size_t i = (size_t)blockIdx.x * 256 + threadIdx.x;
    if (i < n) out[i] = (bf16)in[i];
}

// ---------------------------------------------------------------------------
// V transpose: Vt[(b*NH+h)*DH + d][s] = V[(b*SEQ+s)*DM + h*DH + d]
// ---------------------------------------------------------------------------
__global__ __launch_bounds__(256) void v_transpose_kernel(
    const bf16* __restrict__ V, bf16* __restrict__ Vt, size_t n)
{
    size_t i = (size_t)blockIdx.x * 256 + threadIdx.x;
    if (i >= n) return;
    int s  = (int)(i % SEQ);
    size_t j = i / SEQ;
    int d  = (int)(j % DH);
    int bh = (int)(j / DH);
    int b = bh / NH, h = bh % NH;
    Vt[i] = V[(size_t)(b * SEQ + s) * DM + h * DH + d];
}

// ---------------------------------------------------------------------------
// GEMM: C[M,N] = act( A[M,K] * W[N,K]^T + bias )    (all bf16 in, f32 acc)
// Block: 256 threads = 8 waves, 128x128 tile; wave -> 32 rows x 64 cols.
// Double-buffered LDS, WMMA 16x16x32 bf16.
// ---------------------------------------------------------------------------
#define TM 128
#define TN 128
#define TK 32
#define LDP (TK + 8)   // padded LDS row stride (bf16 elems)

__global__ __launch_bounds__(256) void gemm_bf16_wmma(
    const bf16* __restrict__ A, const bf16* __restrict__ W,
    const float* __restrict__ bias,
    float* __restrict__ outF, bf16* __restrict__ outB,
    int M, int N, int K, int relu)
{
    __shared__ __align__(16) bf16 lA[2][TM][LDP];
    __shared__ __align__(16) bf16 lB[2][TN][LDP];

    const int t    = threadIdx.x;
    const int lane = t & 31;
    const int hi   = (lane >> 4) & 1;
    const int ln   = lane & 15;
    const int wave = t >> 5;
    const int wm   = wave & 3;   // 4 row groups of 32
    const int wn   = wave >> 2;  // 2 col groups of 64

    const int row0 = blockIdx.y * TM;
    const int col0 = blockIdx.x * TN;

    v8f acc[2][4];
    #pragma unroll
    for (int m = 0; m < 2; ++m)
        #pragma unroll
        for (int n = 0; n < 4; ++n)
            #pragma unroll
            for (int r = 0; r < 8; ++r) acc[m][n][r] = 0.f;

    const int KT = K / TK;

    auto loadTile = [&](int kt, int buf) {
        const int k0 = kt * TK;
        #pragma unroll
        for (int i = 0; i < 2; ++i) {
            int lin = t + i * 256;        // 0..511
            int r   = lin >> 2;           // tile row 0..127
            int sg  = lin & 3;            // 16-byte segment within 64B row
            const f32x4* gA = reinterpret_cast<const f32x4*>(
                A + (size_t)(row0 + r) * K + k0 + sg * 8);
            *reinterpret_cast<f32x4*>(&lA[buf][r][sg * 8]) = *gA;
            const f32x4* gB = reinterpret_cast<const f32x4*>(
                W + (size_t)(col0 + r) * K + k0 + sg * 8);
            *reinterpret_cast<f32x4*>(&lB[buf][r][sg * 8]) = *gB;
        }
    };

    loadTile(0, 0);

    const int s0 = hi ? 8 : 0;    // A-fragment K base per half-wave
    const int bo = hi ? 8 : 0;    // B-fragment K base per half-wave

    for (int kt = 0; kt < KT; ++kt) {
        __syncthreads();
        const int cur = kt & 1;
        if (kt + 1 < KT) loadTile(kt + 1, cur ^ 1);
        if (kt + 2 < KT) {
            __builtin_prefetch((const void*)(A + (size_t)(row0 + (t >> 1)) * K + (size_t)(kt + 2) * TK), 0, 1);
            __builtin_prefetch((const void*)(W + (size_t)(col0 + (t >> 1)) * K + (size_t)(kt + 2) * TK), 0, 1);
        }

        Frag af[2];
        #pragma unroll
        for (int m = 0; m < 2; ++m) {
            int rr = wm * 32 + m * 16 + ln;
            af[m].h[0] = *reinterpret_cast<const v8bf*>(&lA[cur][rr][s0]);
            af[m].h[1] = *reinterpret_cast<const v8bf*>(&lA[cur][rr][s0 + 16]);
        }
        Frag bfm[4];
        #pragma unroll
        for (int n = 0; n < 4; ++n) {
            int cc = wn * 64 + n * 16 + ln;
            bfm[n].h[0] = *reinterpret_cast<const v8bf*>(&lB[cur][cc][bo]);
            bfm[n].h[1] = *reinterpret_cast<const v8bf*>(&lB[cur][cc][bo + 16]);
        }
        #pragma unroll
        for (int m = 0; m < 2; ++m)
            #pragma unroll
            for (int n = 0; n < 4; ++n)
                acc[m][n] = wmma_bf16(af[m], bfm[n], acc[m][n]);
    }

    // Epilogue: C frag layout: reg r, lanes0-15 -> (M=r, N=ln); lanes16-31 -> (M=r+8)
    #pragma unroll
    for (int m = 0; m < 2; ++m) {
        #pragma unroll
        for (int n = 0; n < 4; ++n) {
            int gcol = col0 + wn * 64 + n * 16 + ln;
            float bv = bias ? bias[gcol] : 0.f;
            #pragma unroll
            for (int r = 0; r < 8; ++r) {
                int grow = row0 + wm * 32 + m * 16 + r + 8 * hi;
                float v = acc[m][n][r] + bv;
                if (relu) v = fmaxf(v, 0.f);
                size_t off = (size_t)grow * N + gcol;
                if (outF) outF[off] = v;
                else      outB[off] = (bf16)v;
            }
        }
    }
}

// ---------------------------------------------------------------------------
// Flash attention: block = 128 threads (4 waves), each wave owns 16 queries.
// grid = (SEQ/64, NH, BSZ). Online softmax over 32-key blocks.
// ---------------------------------------------------------------------------
__global__ __launch_bounds__(128) void attn_kernel(
    const bf16* __restrict__ Qb, const bf16* __restrict__ Kb,
    const bf16* __restrict__ Vt, const int* __restrict__ mask,
    bf16* __restrict__ Ob)
{
    __shared__ __align__(16) bf16 lP[4][16][40];  // wave-private P tiles

    const int t    = threadIdx.x;
    const int lane = t & 31;
    const int wave = t >> 5;
    const int hi   = (lane >> 4) & 1;
    const int ln   = lane & 15;
    const int h    = blockIdx.y;
    const int b    = blockIdx.z;
    const int q0   = blockIdx.x * 64 + wave * 16;

    const int s0 = hi ? 8 : 0;
    const int ko = hi ? 8 : 0;

    // Q fragments (A-matrix, 16x32 per k-step; d_head=64 -> two frags)
    Frag qf[2];
    {
        size_t rowQ = (size_t)(b * SEQ + q0 + ln) * DM + h * DH;
        #pragma unroll
        for (int kk = 0; kk < 2; ++kk) {
            size_t base = rowQ + kk * 32;
            qf[kk].h[0] = *reinterpret_cast<const v8bf*>(Qb + base + s0);
            qf[kk].h[1] = *reinterpret_cast<const v8bf*>(Qb + base + s0 + 16);
        }
    }

    float mrow[8], lrow[8];
    v8f o[4];
    #pragma unroll
    for (int r = 0; r < 8; ++r) { mrow[r] = -1e30f; lrow[r] = 0.f; }
    #pragma unroll
    for (int n = 0; n < 4; ++n)
        #pragma unroll
        for (int r = 0; r < 8; ++r) o[n][r] = 0.f;

    const bf16* Vbase = Vt + (size_t)(b * NH + h) * DH * SEQ;
    const int*  mrow_mask = mask + (size_t)b * SEQ;

    for (int kb = 0; kb < SEQ; kb += 32) {
        // ---- scores for two 16-key tiles ----
        v8f sc[2];
        #pragma unroll
        for (int kt2 = 0; kt2 < 2; ++kt2) {
            int key = kb + kt2 * 16 + ln;
            size_t rowK = (size_t)(b * SEQ + key) * DM + h * DH;
            Frag kf0, kf1;
            kf0.h[0] = *reinterpret_cast<const v8bf*>(Kb + rowK + ko);
            kf0.h[1] = *reinterpret_cast<const v8bf*>(Kb + rowK + ko + 16);
            kf1.h[0] = *reinterpret_cast<const v8bf*>(Kb + rowK + 32 + ko);
            kf1.h[1] = *reinterpret_cast<const v8bf*>(Kb + rowK + 32 + ko + 16);
            v8f s;
            #pragma unroll
            for (int r = 0; r < 8; ++r) s[r] = 0.f;
            s = wmma_bf16(qf[0], kf0, s);
            s = wmma_bf16(qf[1], kf1, s);
            sc[kt2] = s;
        }
        // scale + mask
        float mA = (mrow_mask[kb + ln]      == 0) ? -1e30f : 0.f;
        float mB = (mrow_mask[kb + 16 + ln] == 0) ? -1e30f : 0.f;
        #pragma unroll
        for (int r = 0; r < 8; ++r) {
            sc[0][r] = sc[0][r] * 0.125f + mA;  // 1/sqrt(64)
            sc[1][r] = sc[1][r] * 0.125f + mB;
        }
        // ---- online softmax stats ----
        float scale[8];
        #pragma unroll
        for (int r = 0; r < 8; ++r) {
            float bm = fmaxf(sc[0][r], sc[1][r]);
            #pragma unroll
            for (int d = 1; d < 16; d <<= 1)
                bm = fmaxf(bm, __shfl_xor(bm, d, 32));
            float mnew = fmaxf(mrow[r], bm);
            scale[r] = __expf(mrow[r] - mnew);
            mrow[r]  = mnew;
        }
        #pragma unroll
        for (int r = 0; r < 8; ++r) {
            float p0 = __expf(sc[0][r] - mrow[r]);
            float p1 = __expf(sc[1][r] - mrow[r]);
            lP[wave][r + 8 * hi][ln]      = (bf16)p0;   // wave-private tile:
            lP[wave][r + 8 * hi][16 + ln] = (bf16)p1;   // same-wave LDS is in-order
            float s2 = p0 + p1;
            #pragma unroll
            for (int d = 1; d < 16; d <<= 1)
                s2 += __shfl_xor(s2, d, 32);
            lrow[r] = lrow[r] * scale[r] + s2;
        }
        // rescale output accumulators
        #pragma unroll
        for (int n = 0; n < 4; ++n)
            #pragma unroll
            for (int r = 0; r < 8; ++r) o[n][r] *= scale[r];

        // ---- read P back in A-fragment layout ----
        Frag pf;
        pf.h[0] = *reinterpret_cast<const v8bf*>(&lP[wave][ln][s0]);
        pf.h[1] = *reinterpret_cast<const v8bf*>(&lP[wave][ln][s0 + 16]);

        // ---- P @ V (V^T gives contiguous B-fragment loads) ----
        #pragma unroll
        for (int n = 0; n < 4; ++n) {
            const bf16* vp = Vbase + (size_t)(n * 16 + ln) * SEQ + kb + ko;
            Frag vf;
            vf.h[0] = *reinterpret_cast<const v8bf*>(vp);
            vf.h[1] = *reinterpret_cast<const v8bf*>(vp + 16);
            o[n] = wmma_bf16(pf, vf, o[n]);
        }
    }

    // ---- finalize and store (concat heads) ----
    #pragma unroll
    for (int n = 0; n < 4; ++n) {
        #pragma unroll
        for (int r = 0; r < 8; ++r) {
            int q = q0 + r + 8 * hi;
            float v = o[n][r] / lrow[r];
            Ob[(size_t)(b * SEQ + q) * DM + h * DH + n * 16 + ln] = (bf16)v;
        }
    }
}

// ---------------------------------------------------------------------------
// Fused residual add + LayerNorm (ref: (x-mu)/(std+eps)*g + b), D = 1024
// ---------------------------------------------------------------------------
__global__ __launch_bounds__(256) void add_ln_kernel(
    const float* __restrict__ A, const float* __restrict__ Badd,
    const float* __restrict__ gam, const float* __restrict__ bet,
    float* __restrict__ outF, bf16* __restrict__ outB)
{
    __shared__ float red[256];
    const int row = blockIdx.x, t = threadIdx.x;
    const float* a = A    + (size_t)row * DM;
    const float* b = Badd + (size_t)row * DM;

    float v[4]; float s = 0.f;
    #pragma unroll
    for (int i = 0; i < 4; ++i) {
        int c = t + i * 256;
        float x = a[c] + b[c];
        v[i] = x; s += x;
    }
    red[t] = s; __syncthreads();
    for (int st = 128; st > 0; st >>= 1) {
        if (t < st) red[t] += red[t + st];
        __syncthreads();
    }
    float mu = red[0] * (1.f / DM);
    __syncthreads();

    float vs = 0.f;
    #pragma unroll
    for (int i = 0; i < 4; ++i) { float d = v[i] - mu; vs += d * d; }
    red[t] = vs; __syncthreads();
    for (int st = 128; st > 0; st >>= 1) {
        if (t < st) red[t] += red[t + st];
        __syncthreads();
    }
    float inv = 1.f / (sqrtf(red[0] * (1.f / DM)) + 1e-6f);

    #pragma unroll
    for (int i = 0; i < 4; ++i) {
        int c = t + i * 256;
        float y = (v[i] - mu) * inv * gam[c] + bet[c];
        outF[(size_t)row * DM + c] = y;
        if (outB) outB[(size_t)row * DM + c] = (bf16)y;
    }
}

// ---------------------------------------------------------------------------
// Host orchestration
// ---------------------------------------------------------------------------
extern "C" void kernel_launch(void* const* d_in, const int* in_sizes, int n_in,
                              void* d_out, int out_size, void* d_ws, size_t ws_size,
                              hipStream_t stream) {
    (void)in_sizes; (void)n_in; (void)out_size; (void)ws_size;
    const float* x    = (const float*)d_in[0];
    const int*   mask = (const int*)  d_in[1];
    const float* Wq   = (const float*)d_in[2];
    const float* Wk   = (const float*)d_in[3];
    const float* Wv   = (const float*)d_in[4];
    const float* Wo   = (const float*)d_in[5];
    const float* w1   = (const float*)d_in[6];
    const float* b1   = (const float*)d_in[7];
    const float* w2   = (const float*)d_in[8];
    const float* b2   = (const float*)d_in[9];
    const float* g1   = (const float*)d_in[10];
    const float* be1  = (const float*)d_in[11];
    const float* g2   = (const float*)d_in[12];
    const float* be2  = (const float*)d_in[13];

    // workspace carve-out (256B aligned)
    size_t off = 0;
    auto alloc = [&](size_t bytes) {
        void* p = (char*)d_ws + off;
        off += (bytes + 255) & ~(size_t)255;
        return p;
    };
    bf16* xb   = (bf16*)alloc((size_t)MR * DM * 2);
    bf16* wqb  = (bf16*)alloc((size_t)DM * DM * 2);
    bf16* wkb  = (bf16*)alloc((size_t)DM * DM * 2);
    bf16* wvb  = (bf16*)alloc((size_t)DM * DM * 2);
    bf16* wob  = (bf16*)alloc((size_t)DM * DM * 2);
    bf16* w1b  = (bf16*)alloc((size_t)DFF * DM * 2);
    bf16* w2b  = (bf16*)alloc((size_t)DM * DFF * 2);
    bf16* Qb   = (bf16*)alloc((size_t)MR * DM * 2);
    bf16* Kb   = (bf16*)alloc((size_t)MR * DM * 2);
    bf16* Vb   = (bf16*)alloc((size_t)MR * DM * 2);
    bf16* Vt   = (bf16*)alloc((size_t)MR * DM * 2);
    bf16* Ab   = (bf16*)alloc((size_t)MR * DM * 2);   // attention output (bf16)
    float* x1  = (float*)alloc((size_t)MR * DM * 4);
    float* x2f = (float*)alloc((size_t)MR * DM * 4);
    bf16*  x2b = (bf16*) alloc((size_t)MR * DM * 2);
    bf16*  hb  = (bf16*) alloc((size_t)MR * DFF * 2);
    float* x3  = (float*)alloc((size_t)MR * DM * 4);

    auto cvt = [&](const float* src, bf16* dst, size_t n) {
        f32_to_bf16_kernel<<<dim3((unsigned)((n + 255) / 256)), 256, 0, stream>>>(src, dst, n);
    };
    cvt(x,  xb,  (size_t)MR * DM);
    cvt(Wq, wqb, (size_t)DM * DM);
    cvt(Wk, wkb, (size_t)DM * DM);
    cvt(Wv, wvb, (size_t)DM * DM);
    cvt(Wo, wob, (size_t)DM * DM);
    cvt(w1, w1b, (size_t)DFF * DM);
    cvt(w2, w2b, (size_t)DM * DFF);

    // Q/K/V projections: [MR,DM] x [DM,DM]^T -> bf16
    dim3 gqkv(DM / TN, MR / TM);
    gemm_bf16_wmma<<<gqkv, 256, 0, stream>>>(xb, wqb, nullptr, nullptr, Qb, MR, DM, DM, 0);
    gemm_bf16_wmma<<<gqkv, 256, 0, stream>>>(xb, wkb, nullptr, nullptr, Kb, MR, DM, DM, 0);
    gemm_bf16_wmma<<<gqkv, 256, 0, stream>>>(xb, wvb, nullptr, nullptr, Vb, MR, DM, DM, 0);

    // V -> V^T per head
    {
        size_t n = (size_t)BSZ * NH * DH * SEQ;
        v_transpose_kernel<<<dim3((unsigned)((n + 255) / 256)), 256, 0, stream>>>(Vb, Vt, n);
    }

    // attention
    attn_kernel<<<dim3(SEQ / 64, NH, BSZ), 128, 0, stream>>>(Qb, Kb, Vt, mask, Ab);

    // output projection -> x1 (f32)
    gemm_bf16_wmma<<<gqkv, 256, 0, stream>>>(Ab, wob, nullptr, x1, nullptr, MR, DM, DM, 0);

    // x2 = LN(x1 + x)  (f32 + bf16 copies)
    add_ln_kernel<<<MR, 256, 0, stream>>>(x1, x, g1, be1, x2f, x2b);

    // FFN1: relu(x2 @ w1^T + b1) -> bf16 [MR, DFF]
    gemm_bf16_wmma<<<dim3(DFF / TN, MR / TM), 256, 0, stream>>>(x2b, w1b, b1, nullptr, hb, MR, DFF, DM, 1);

    // FFN2: h @ w2^T + b2 -> f32 [MR, DM]
    gemm_bf16_wmma<<<dim3(DM / TN, MR / TM), 256, 0, stream>>>(hb, w2b, b2, x3, nullptr, MR, DM, DFF, 0);

    // out = LN(x2 + x3)
    add_ln_kernel<<<MR, 256, 0, stream>>>(x2f, x3, g2, be2, (float*)d_out, nullptr);
}
